// AttentionBlock_10703058502010
// MI455X (gfx1250) — compile-verified, matrix operand order checked by reading, not run on hardware
//
#include <hip/hip_runtime.h>
#include <hip/hip_bf16.h>
#include <stdint.h>

// CDNA5 / gfx1250, wave32. bf16 WMMA (v_wmma_f32_16x16x32_bf16) with f32 accum.
// Flash-style fused softmax(leaky_relu(s1_i+s2_j)) @ h using the exact row max
// m_i = lrelu(s1_i + max_j s2_j) (lrelu monotonic => no online rescale).
// B-operand tiles (weights / V) are staged into LDS by the Tensor Data Mover
// (TENSOR_LOAD_TO_LDS, TENSORcnt, double buffered); waves consume them as 16B
// ds_load_b128 fragments. TDM issue is guarded by a *scalar-uniform* branch
// (readfirstlane) since TDM ignores EXEC -- an exec-masked guard would issue
// duplicate DMAs from every wave.

typedef __bf16 bf16;
typedef bf16  v16bf __attribute__((ext_vector_type(16)));
typedef float v8f   __attribute__((ext_vector_type(8)));
typedef unsigned int v4u __attribute__((ext_vector_type(4)));

union Frag { v16bf bf; v4u u[2]; };

__device__ __forceinline__ v4u ld128(const bf16* p) {
  return *reinterpret_cast<const v4u*>(p);
}

__device__ __forceinline__ float lrelu01(float x) { return x > 0.f ? x : 0.01f * x; }

// wave index as an SGPR value -> scalar branches around TDM code
__device__ __forceinline__ int uniform_wave_id() {
  return __builtin_amdgcn_readfirstlane((int)threadIdx.x) >> 5;
}

// ---------------------------------------------------------------------------
// Tensor Data Mover: 2-D tile (tile_d0 contiguous elements x tile_d1 rows of a
// row-major tensor with leading-dim stride_elems) -> LDS. bf16 (data_size=2B).
// Descriptor bit layout per CDNA5 ISA sec 8.3/8.4 (D# group0/group1).
// ---------------------------------------------------------------------------
#if defined(__HIP_DEVICE_COMPILE__) && \
    __has_builtin(__builtin_amdgcn_tensor_load_to_lds) && \
    __has_builtin(__builtin_amdgcn_s_wait_tensorcnt)
#define HAVE_TDM 1
#else
#define HAVE_TDM 0
#endif

#if HAVE_TDM
typedef int tdm_v4i __attribute__((ext_vector_type(4)));
typedef int tdm_v8i __attribute__((ext_vector_type(8)));

__device__ __forceinline__ void tdm_load_2d(
    const void* tile_gaddr, void* lds_dst,
    unsigned tile_d0, unsigned tile_d1,
    unsigned tensor_d0, unsigned tensor_d1,
    unsigned long long stride_elems)
{
  const unsigned long long ga = (unsigned long long)(uintptr_t)tile_gaddr;
  const unsigned lds = (unsigned)(uintptr_t)lds_dst;  // low 32 bits = LDS offset

  v4u g0;
  g0.x = 1u;                                      // count=1 (valid), user mode
  g0.y = lds;                                     // lds_addr
  g0.z = (unsigned)(ga & 0xFFFFFFFFull);          // global_addr[31:0]
  g0.w = (unsigned)((ga >> 32) & 0x01FFFFFFull)   // global_addr[56:32]
         | (2u << 30);                            // type=2 ("image")

  tdm_v8i g1;
  g1[0] = (int)(1u << 16);                        // data_size=1 -> 2 bytes; no mask
  g1[1] = (int)((tensor_d0 & 0xFFFFu) << 16);     // tensor_dim0 lo at bits 63:48
  g1[2] = (int)(((tensor_d0 >> 16) & 0xFFFFu) | ((tensor_d1 & 0xFFFFu) << 16));
  g1[3] = (int)(((tensor_d1 >> 16) & 0xFFFFu) | ((tile_d0 & 0xFFFFu) << 16));
  g1[4] = (int)(tile_d1 & 0xFFFFu);               // tile_dim1 (tile_dim2 = 0)
  g1[5] = (int)(unsigned)(stride_elems & 0xFFFFFFFFull);      // dim0 stride lo
  g1[6] = (int)(unsigned)((stride_elems >> 32) & 0xFFFFull);  // dim0 stride hi
  g1[7] = 0;                                      // dim1 stride unused (2-D)

  tdm_v4i g2 = {0, 0, 0, 0};
  tdm_v4i g3 = {0, 0, 0, 0};
#if defined(__clang_major__) && (__clang_major__ >= 23)
  tdm_v8i g4 = {0, 0, 0, 0, 0, 0, 0, 0};
  __builtin_amdgcn_tensor_load_to_lds(g0, g1, g2, g3, g4, 0);
#else
  __builtin_amdgcn_tensor_load_to_lds(g0, g1, g2, g3, 0);
#endif
}
#endif  // HAVE_TDM

// ---------------------------------------------------------------------------
// Generic bf16 GEMM:  D[M x N] = A[M x K] (row major) * Bt[N x K] (row major,
// i.e. B^T with K contiguous -- torch (out,in) weight layout). N == 512.
// Block: 16 rows x 512 cols; 8 waves x 4 N-tiles each.
// B tiles (512 x 32 k-chunk, 32KB) are TDM-DMA'd into LDS, double buffered.
// All four B fragments are loaded before the WMMAs so the LDS latency is
// covered by partial s_wait_dscnt waits instead of full serialization.
// EPI 0: outf = acc + xin + bias
// EPI 1: outT[col*nrows+row] = bf16(acc)   (h stored transposed for attention)
// EPI 2: outf += elu(acc) * scale
// ---------------------------------------------------------------------------
template <int EPI>
__global__ __launch_bounds__(256) void k_gemm16(
    const bf16* __restrict__ A, const bf16* __restrict__ Bt, int Kdim, int Ncols,
    const float* __restrict__ xin, const float* __restrict__ bias,
    float* __restrict__ outf, bf16* __restrict__ outT, int nrows, float scale)
{
  const int tid  = threadIdx.x;
  const int wave = tid >> 5, lane = tid & 31;
  const int m    = lane & 15;
  const int half = lane >> 4;
  const int kb   = half * 8;
  const int row0 = blockIdx.x * 16;
  const int c0   = wave * 64;

  const bf16* ap = A + (size_t)(row0 + m) * Kdim;
  v8f acc[4] = {};

#if HAVE_TDM
  __shared__ alignas(16) bf16 Bl[2][512 * 32];
  const int nch = Kdim / 32;
  const int uw  = uniform_wave_id();
  if (uw == 0)   // scalar branch: only wave 0 issues the DMA (TDM ignores EXEC)
    tdm_load_2d(Bt, Bl[0], 32u, (unsigned)Ncols, (unsigned)Kdim, (unsigned)Ncols,
                (unsigned long long)Kdim);
  for (int it = 0; it < nch; ++it) {
    const int kc = it * 32;
    Frag a;
    a.u[0] = ld128(ap + kc + kb);
    a.u[1] = ld128(ap + kc + kb + 16);
    __builtin_prefetch(ap + kc + 32, 0, 1);  // next k-chunk of A
    if (uw == 0) __builtin_amdgcn_s_wait_tensorcnt(0);
    __syncthreads();
    if (uw == 0 && it + 1 < nch)
      tdm_load_2d(Bt + (size_t)(it + 1) * 32, Bl[(it + 1) & 1], 32u,
                  (unsigned)Ncols, (unsigned)Kdim, (unsigned)Ncols,
                  (unsigned long long)Kdim);
    const bf16* Bw = Bl[it & 1];
    Frag b[4];
#pragma unroll
    for (int t = 0; t < 4; ++t) {
      const bf16* bp = Bw + (c0 + t * 16 + m) * 32;
      b[t].u[0] = *reinterpret_cast<const v4u*>(bp + kb);
      b[t].u[1] = *reinterpret_cast<const v4u*>(bp + kb + 16);
    }
#pragma unroll
    for (int t = 0; t < 4; ++t)
      acc[t] = __builtin_amdgcn_wmma_f32_16x16x32_bf16(
          false, a.bf, false, b[t].bf, (short)0, acc[t], false, false);
  }
#else
  for (int kc = 0; kc < Kdim; kc += 32) {
    Frag a;
    a.u[0] = ld128(ap + kc + kb);
    a.u[1] = ld128(ap + kc + kb + 16);
    Frag b[4];
#pragma unroll
    for (int t = 0; t < 4; ++t) {
      const bf16* bp = Bt + (size_t)(c0 + t * 16 + m) * Kdim + kc;
      b[t].u[0] = ld128(bp + kb);
      b[t].u[1] = ld128(bp + kb + 16);
    }
#pragma unroll
    for (int t = 0; t < 4; ++t)
      acc[t] = __builtin_amdgcn_wmma_f32_16x16x32_bf16(
          false, a.bf, false, b[t].bf, (short)0, acc[t], false, false);
  }
#endif

#pragma unroll
  for (int t = 0; t < 4; ++t) {
#pragma unroll
    for (int r = 0; r < 8; ++r) {
      const int row = row0 + r + 8 * half;
      const int col = c0 + t * 16 + m;
      const float v = acc[t][r];
      if (EPI == 0) {
        const size_t o = (size_t)row * Ncols + col;
        outf[o] = v + xin[o] + bias[col];
      } else if (EPI == 1) {
        outT[(size_t)col * nrows + row] = (bf16)v;
      } else {
        const size_t o = (size_t)row * Ncols + col;
        const float e = v > 0.f ? v : (__expf(v) - 1.f);  // elu
        outf[o] += e * scale;
      }
    }
  }
}

// ---------------------------------------------------------------------------
// Fused attention: out[i,:] = (1/l_i) * sum_j exp(lrelu(s1_i+s2_j) - m_i) * h[j,:]
// Block = 16 query rows x 512 cols (8 waves x 64 cols).
// P tile (16x32 bf16) produced cooperatively into LDS (double buffered);
// V tile (512x32 from hT) DMA'd into LDS by TDM (double buffered).
// One barrier per j-chunk covers both buffers.
// ---------------------------------------------------------------------------
__global__ __launch_bounds__(256) void k_attn(
    const bf16* __restrict__ hT, const float* __restrict__ s1,
    const float* __restrict__ s2, const float* __restrict__ s2mx_p,
    float* __restrict__ outf, int n, int Ncols)
{
  __shared__ alignas(16) bf16 Pbuf[2][16 * 32];
  __shared__ float rowsum[16];

  const int tid  = threadIdx.x;
  const int wave = tid >> 5, lane = tid & 31;
  const int m    = lane & 15;
  const int half = lane >> 4;
  const int kb   = half * 8;
  const int row0 = blockIdx.x * 16;
  const int c0   = wave * 64;
  const float s2mx = s2mx_p[0];

  const int pcol = lane;        // producer: column within chunk
  const int pr   = wave * 2;    // producer: wave w makes rows 2w, 2w+1
  const float s1a = s1[row0 + pr];
  const float s1b = s1[row0 + pr + 1];
  const float ma  = lrelu01(s1a + s2mx);  // exact row max
  const float mb  = lrelu01(s1b + s2mx);
  float la = 0.f, lb = 0.f;

  v8f acc[4] = {};
  const int nch = n / 32;

#if HAVE_TDM
  __shared__ alignas(16) bf16 Vbuf[2][512 * 32];
  const int uw = uniform_wave_id();
  if (uw == 0)
    tdm_load_2d(hT, Vbuf[0], 32u, 512u, (unsigned)n, (unsigned)512,
                (unsigned long long)n);
#endif

  for (int it = 0; it < nch; ++it) {
    const int jc = it * 32;
    bf16* P = Pbuf[it & 1];
    const float sv = s2[jc + pcol];
    const float pa = __expf(lrelu01(s1a + sv) - ma);
    const float pb = __expf(lrelu01(s1b + sv) - mb);
    la += pa;
    lb += pb;
    P[pr * 32 + pcol]       = (bf16)pa;
    P[(pr + 1) * 32 + pcol] = (bf16)pb;
#if HAVE_TDM
    if (uw == 0) __builtin_amdgcn_s_wait_tensorcnt(0);
#endif
    __syncthreads();
#if HAVE_TDM
    if (uw == 0 && it + 1 < nch)
      tdm_load_2d(hT + (size_t)(it + 1) * 32, Vbuf[(it + 1) & 1], 32u, 512u,
                  (unsigned)n, (unsigned)512, (unsigned long long)n);
#endif

    Frag a;
    a.u[0] = *reinterpret_cast<const v4u*>(&P[m * 32 + kb]);
    a.u[1] = *reinterpret_cast<const v4u*>(&P[m * 32 + kb + 16]);

#if HAVE_TDM
    const bf16* Vb = Vbuf[it & 1];
    Frag b[4];
#pragma unroll
    for (int t = 0; t < 4; ++t) {
      const bf16* bp = Vb + (c0 + t * 16 + m) * 32;
      b[t].u[0] = *reinterpret_cast<const v4u*>(bp + kb);
      b[t].u[1] = *reinterpret_cast<const v4u*>(bp + kb + 16);
    }
#pragma unroll
    for (int t = 0; t < 4; ++t)
      acc[t] = __builtin_amdgcn_wmma_f32_16x16x32_bf16(
          false, a.bf, false, b[t].bf, (short)0, acc[t], false, false);
#else
    Frag b[4];
#pragma unroll
    for (int t = 0; t < 4; ++t) {
      const bf16* bp = hT + (size_t)(c0 + t * 16 + m) * n + jc;
      __builtin_prefetch(bp + 32, 0, 1);
      b[t].u[0] = ld128(bp + kb);
      b[t].u[1] = ld128(bp + kb + 16);
    }
#pragma unroll
    for (int t = 0; t < 4; ++t)
      acc[t] = __builtin_amdgcn_wmma_f32_16x16x32_bf16(
          false, a.bf, false, b[t].bf, (short)0, acc[t], false, false);
#endif
  }

  // row sums: reduce producer partials across the wave's 32 column-lanes
#pragma unroll
  for (int s = 16; s > 0; s >>= 1) {
    la += __shfl_xor(la, s, 32);
    lb += __shfl_xor(lb, s, 32);
  }
  if (lane == 0) { rowsum[pr] = la; rowsum[pr + 1] = lb; }
  __syncthreads();

  float linv[8];
#pragma unroll
  for (int r = 0; r < 8; ++r) linv[r] = 1.0f / rowsum[r + 8 * half];

#pragma unroll
  for (int t = 0; t < 4; ++t)
#pragma unroll
    for (int r = 0; r < 8; ++r) {
      const int row = row0 + r + 8 * half;
      const int col = c0 + t * 16 + m;
      outf[(size_t)row * Ncols + col] = acc[t][r] * linv[r];
    }
}

// ---------------------------------------------------------------------------
// s1[r] = h[r,:].a1, s2[r] = h[r,:].a2 from hT (coalesced over r), + block max s2
// ---------------------------------------------------------------------------
__global__ __launch_bounds__(256) void k_rowdot(
    const bf16* __restrict__ hT, const float* __restrict__ Wa, int C, int n,
    float* __restrict__ s1, float* __restrict__ s2, float* __restrict__ pmax)
{
  const int r = blockIdx.x * 256 + threadIdx.x;
  float a = 0.f, b = 0.f;
  for (int c = 0; c < C; ++c) {
    const float hv = (float)hT[(size_t)c * n + r];
    a = fmaf(hv, Wa[c], a);
    b = fmaf(hv, Wa[C + c], b);
  }
  s1[r] = a;
  s2[r] = b;
  float mx = b;
#pragma unroll
  for (int s = 16; s > 0; s >>= 1) mx = fmaxf(mx, __shfl_xor(mx, s, 32));
  __shared__ float wm[8];
  if ((threadIdx.x & 31) == 0) wm[threadIdx.x >> 5] = mx;
  __syncthreads();
  if (threadIdx.x == 0) {
    float t = wm[0];
#pragma unroll
    for (int i = 1; i < 8; ++i) t = fmaxf(t, wm[i]);
    pmax[blockIdx.x] = t;
  }
}

__global__ void k_maxred(const float* __restrict__ pmax, int nb, float* __restrict__ out)
{
  float mx = -3.4e38f;
  for (int i = threadIdx.x; i < nb; i += 32) mx = fmaxf(mx, pmax[i]);
#pragma unroll
  for (int s = 16; s > 0; s >>= 1) mx = fmaxf(mx, __shfl_xor(mx, s, 32));
  if (threadIdx.x == 0) out[0] = mx;
}

// ---------------------------------------------------------------------------
// LayerNorm over C=512 (block per row, 2 elems/thread), optional residual,
// bf16 output (A operand for the next WMMA GEMM).
// ---------------------------------------------------------------------------
__global__ __launch_bounds__(256) void k_layernorm(
    const float* __restrict__ in, const float* __restrict__ resid,
    const float* __restrict__ g, const float* __restrict__ b,
    bf16* __restrict__ out, int C)
{
  const int row = blockIdx.x, tid = threadIdx.x;
  const size_t base = (size_t)row * C;
  float v0 = in[base + tid];
  float v1 = in[base + 256 + tid];
  if (resid) { v0 += resid[base + tid]; v1 += resid[base + 256 + tid]; }

  float s = v0 + v1;
#pragma unroll
  for (int sh = 16; sh > 0; sh >>= 1) s += __shfl_xor(s, sh, 32);
  __shared__ float w1[8];
  if ((tid & 31) == 0) w1[tid >> 5] = s;
  __syncthreads();
  float tot = 0.f;
#pragma unroll
  for (int i = 0; i < 8; ++i) tot += w1[i];
  const float mu = tot / (float)C;

  const float d0 = v0 - mu, d1 = v1 - mu;
  float q = d0 * d0 + d1 * d1;
#pragma unroll
  for (int sh = 16; sh > 0; sh >>= 1) q += __shfl_xor(q, sh, 32);
  __shared__ float w2[8];
  if ((tid & 31) == 0) w2[tid >> 5] = q;
  __syncthreads();
  float qt = 0.f;
#pragma unroll
  for (int i = 0; i < 8; ++i) qt += w2[i];
  const float rinv = rsqrtf(qt / (float)C + 1e-5f);

  out[base + tid]       = (bf16)(d0 * rinv * g[tid] + b[tid]);
  out[base + 256 + tid] = (bf16)(d1 * rinv * g[256 + tid] + b[256 + tid]);
}

__global__ void k_f2bf(const float* __restrict__ in, bf16* __restrict__ out, size_t cnt)
{
  const size_t i = (size_t)blockIdx.x * blockDim.x + threadIdx.x;
  if (i < cnt) out[i] = (bf16)in[i];
}

__global__ void k_zero(float* __restrict__ p, size_t cnt)
{
  const size_t i = (size_t)blockIdx.x * blockDim.x + threadIdx.x;
  if (i < cnt) p[i] = 0.f;
}

// ---------------------------------------------------------------------------
extern "C" void kernel_launch(void* const* d_in, const int* in_sizes, int n_in,
                              void* d_out, int out_size, void* d_ws, size_t ws_size,
                              hipStream_t stream)
{
  const float* x    = (const float*)d_in[0];
  const float* enc  = (const float*)d_in[1];
  const float* Wenc = (const float*)d_in[2];
  const float* benc = (const float*)d_in[3];
  const float* ln1g = (const float*)d_in[4];
  const float* ln1b = (const float*)d_in[5];
  const float* ln2g = (const float*)d_in[6];
  const float* ln2b = (const float*)d_in[7];
  const float* W0   = (const float*)d_in[8];
  const float* W1   = (const float*)d_in[9];
  const float* Wa   = (const float*)d_in[10];

  const int C    = in_sizes[3];              // 512
  const int n    = in_sizes[0] / C;          // 8192
  const int E    = in_sizes[2] / C;          // 256
  const int K    = in_sizes[8] / (C * C);    // 2
  const int Cout = in_sizes[9] / (K * C);    // 512
  float* hnext = (float*)d_out;

  // Workspace carve (~62.5 MB; working set sits inside the 192MB L2).
  char* w = (char*)d_ws;
  auto carve = [&](size_t bytes) -> char* {
    char* p = w;
    w += (bytes + 255) & ~(size_t)255;
    return p;
  };
  bf16*  enc_bf  = (bf16*)carve((size_t)n * E * 2);
  bf16*  Wenc_bf = (bf16*)carve((size_t)C * E * 2);
  bf16*  W0_bf   = (bf16*)carve((size_t)K * C * C * 2);
  bf16*  W1_bf   = (bf16*)carve((size_t)K * Cout * C * 2);
  float* x_f     = (float*)carve((size_t)n * C * 4);
  bf16*  xn_bf   = (bf16*)carve((size_t)n * C * 2);
  bf16*  hT      = (bf16*)carve((size_t)C * n * 2);
  float* attn_f  = (float*)carve((size_t)n * C * 4);
  bf16*  y_bf    = (bf16*)carve((size_t)n * C * 2);
  float* s1      = (float*)carve((size_t)n * 4);
  float* s2      = (float*)carve((size_t)n * 4);
  float* pmax    = (float*)carve(4096);
  float* s2mx    = (float*)carve(256);

  auto cvt = [&](const float* src, bf16* dst, size_t cnt) {
    k_f2bf<<<(unsigned)((cnt + 255) / 256), 256, 0, stream>>>(src, dst, cnt);
  };
  cvt(enc, enc_bf, (size_t)n * E);
  cvt(Wenc, Wenc_bf, (size_t)C * E);
  cvt(W0, W0_bf, (size_t)K * C * C);
  cvt(W1, W1_bf, (size_t)K * Cout * C);

  const dim3 blk(256);
  const int rowBlocks = n / 16;

  // x_f = enc @ Wenc^T + x + b_enc
  k_gemm16<0><<<rowBlocks, blk, 0, stream>>>(enc_bf, Wenc_bf, E, C, x, benc,
                                             x_f, nullptr, n, 0.f);
  k_layernorm<<<n, blk, 0, stream>>>(x_f, nullptr, ln1g, ln1b, xn_bf, C);
  k_zero<<<(unsigned)(((size_t)n * Cout + 255) / 256), 256, 0, stream>>>(
      hnext, (size_t)n * Cout);

  const float kscale = 1.0f / (float)K;
  for (int h = 0; h < K; ++h) {
    // h = x_norm @ W0[h]^T, stored transposed (C x n) as attention V
    k_gemm16<1><<<rowBlocks, blk, 0, stream>>>(xn_bf, W0_bf + (size_t)h * C * C,
                                               C, C, nullptr, nullptr, nullptr,
                                               hT, n, 0.f);
    k_rowdot<<<n / 256, blk, 0, stream>>>(hT, Wa + (size_t)h * 2 * C, C, n,
                                          s1, s2, pmax);
    k_maxred<<<1, 32, 0, stream>>>(pmax, n / 256, s2mx);
    // fused softmax(lrelu(s1_i+s2_j)) @ h
    k_attn<<<rowBlocks, blk, 0, stream>>>(hT, s1, s2, s2mx, attn_f, n, C);
    // residual + LN2 -> bf16
    k_layernorm<<<n, blk, 0, stream>>>(attn_f, x_f, ln2g, ln2b, y_bf, C);
    // h_next += elu(y @ W1[h]^T) / K
    k_gemm16<2><<<rowBlocks, blk, 0, stream>>>(y_bf, W1_bf + (size_t)h * Cout * C,
                                               C, Cout, nullptr, nullptr, hnext,
                                               nullptr, n, kscale);
  }
}
